// ATTLSTMcell_12532714569752
// MI455X (gfx1250) — matrix operand order repaired and checked by reading, not compiled, not run on hardware
//
#include <hip/hip_runtime.h>
#include <hip/hip_bf16.h>

// ---------------- problem constants ----------------
constexpr int B_   = 32;
constexpr int T_   = 512;
constexpr int H_   = 1024;
constexpr int N_   = 512;
constexpr int K3H_ = 3072;   // 3*H

// ---------------- WMMA types ----------------
typedef __attribute__((ext_vector_type(16))) __bf16 v16bf;
typedef __attribute__((ext_vector_type(8)))  float  v8f;

union Frag16 { v16bf bf; uint4 u[2]; };

// 16-bit A-matrix 16x32 (MxK), ISA 7.12.2:
// lanes 0-15 : elems 0-7 = K 0..7,   elems 8-15 = K 16..23   (row M = lane)
// lanes 16-31: elems 0-7 = K 8..15,  elems 8-15 = K 24..31   (row M = lane-16)
__device__ inline v16bf load_a16(const __bf16* __restrict__ base, int stride,
                                 int m0, int kk, int lane) {
  const int r = lane & 15, half = lane >> 4;
  const __bf16* p = base + (size_t)(m0 + r) * stride + kk + half * 8;
  Frag16 f;
  f.u[0] = *(const uint4*)(p);
  f.u[1] = *(const uint4*)(p + 16);
  return f.bf;
}

// 16-bit B-matrix 32x16 (KxN), same K order as A (column N = lane%16):
// lanes 0-15 hold K 0..7,16..23; lanes 16-31 hold K 8..15,24..31.
// `base` is B^T stored [N, K] row-major -> two 16-byte chunks per lane.
__device__ inline v16bf load_b16(const __bf16* __restrict__ base, int stride,
                                 int n0, int kk, int lane) {
  const int c = lane & 15, half = lane >> 4;
  const __bf16* p = base + (size_t)(n0 + c) * stride + kk + half * 8;
  Frag16 f;
  f.u[0] = *(const uint4*)(p);
  f.u[1] = *(const uint4*)(p + 16);
  return f.bf;
}

__device__ inline v8f wmma_bf16(v16bf a, v16bf b, v8f c) {
  return __builtin_amdgcn_wmma_f32_16x16x32_bf16(false, a, false, b,
                                                 (short)0, c, false, false);
}

__device__ inline float hsig(float v) {
  return fminf(fmaxf(0.2f * v + 0.5f, 0.0f), 1.0f);
}

// ---------------- prologue kernels ----------------
__global__ __launch_bounds__(256) void cvt_bf16_k(const float* __restrict__ in,
                                                  __bf16* __restrict__ out, int n) {
  int i = blockIdx.x * 256 + threadIdx.x;
  if (i < n) out[i] = (__bf16)in[i];
}

__global__ __launch_bounds__(256) void copy_f32_k(const float* __restrict__ in,
                                                  float* __restrict__ out, int n) {
  int i = blockIdx.x * 256 + threadIdx.x;
  if (i < n) out[i] = in[i];
}

// out[c][r] = (bf16) in[r][c]; in is [rows, cols] fp32, out is [cols, rows] bf16
__global__ __launch_bounds__(256) void transpose_cvt_k(const float* __restrict__ in,
                                                       __bf16* __restrict__ out,
                                                       int rows, int cols) {
  __shared__ float tile[32][33];
  const int bx = blockIdx.x * 32;   // col base
  const int by = blockIdx.y * 32;   // row base
  const int tx = threadIdx.x, ty = threadIdx.y;
  #pragma unroll
  for (int i = 0; i < 32; i += 8)
    tile[ty + i][tx] = in[(size_t)(by + ty + i) * cols + bx + tx];
  __syncthreads();
  #pragma unroll
  for (int i = 0; i < 32; i += 8)
    out[(size_t)(bx + ty + i) * rows + by + tx] = (__bf16)tile[tx][ty + i];
}

// ---------------- K1: fused gates GEMM + LSTM cell update ----------------
// grid: 64 blocks (one 16-wide H tile each), 128 threads = 4 waves.
// K split: wave w handles k-steps  rg*32 + w*8 + j  (rg in {0,1,2}, j in [0,8)).
// The rg loop is fully unrolled so the pack-region base/stride selection is
// compile-time static -> no divergent control flow in the hot loop.
__global__ __launch_bounds__(128) void k1_gates(
    const __bf16* __restrict__ wt,     // [4][H][3H] bf16 (pre-transposed)
    const __bf16* __restrict__ hprev,  // [B,H] bf16
    const __bf16* __restrict__ rprev,  // [B,H] bf16
    const __bf16* __restrict__ xbf,    // [B,T,H] bf16
    const float*  __restrict__ x,      // [B,T,H] fp32 (residual)
    const float*  __restrict__ bI, const float* __restrict__ bF,
    const float*  __restrict__ bC, const float* __restrict__ bO,
    float*  __restrict__ cbuf,         // [B,H] fp32, updated in place
    float*  __restrict__ out,          // [B,T,H] fp32
    __bf16* __restrict__ hcur,         // [B,H] bf16
    int t) {
  __shared__ float red[4][8][8][32];
  const int nt   = blockIdx.x;          // 0..63  -> H tile
  const int w    = threadIdx.x >> 5;    // wave 0..3 -> K split within region
  const int lane = threadIdx.x & 31;

  v8f acc[8] = {};                      // [gate(4)][mt(2)]

  const int kw = w * 256;               // this wave's k offset inside a region
  #pragma unroll
  for (int rg = 0; rg < 3; ++rg) {
    // compile-time region selection (rg is an unrolled constant)
    const __bf16* abase;
    int astride;
    if (rg == 0)      { abase = hprev;                astride = H_;      }
    else if (rg == 1) { abase = rprev;                astride = H_;      }
    else              { abase = xbf + (size_t)t * H_; astride = T_ * H_; }

    for (int j = 0; j < 8; ++j) {
      const int kk = kw + j * 32;       // k offset within region [0, 1024)
      v16bf a0 = load_a16(abase, astride, 0,  kk, lane);
      v16bf a1 = load_a16(abase, astride, 16, kk, lane);
      const int kglob = rg * 1024 + kk; // k offset within full 3H pack
      #pragma unroll
      for (int g = 0; g < 4; ++g) {
        v16bf bb = load_b16(wt + (size_t)g * H_ * K3H_, K3H_, nt * 16, kglob, lane);
        acc[g * 2 + 0] = wmma_bf16(a0, bb, acc[g * 2 + 0]);
        acc[g * 2 + 1] = wmma_bf16(a1, bb, acc[g * 2 + 1]);
      }
    }
  }

  #pragma unroll
  for (int tl = 0; tl < 8; ++tl)
    #pragma unroll
    for (int v = 0; v < 8; ++v)
      red[w][tl][v][lane] = acc[tl][v];
  __syncthreads();

  if (w < 2) {
    const int mt   = w;
    const int half = lane >> 4, cc = lane & 15;
    const int h    = nt * 16 + cc;
    const float biI = bI[h], biF = bF[h], biC = bC[h], biO = bO[h];
    #pragma unroll
    for (int v = 0; v < 8; ++v) {
      const int b = mt * 16 + v + 8 * half;
      float s[4];
      #pragma unroll
      for (int g = 0; g < 4; ++g)
        s[g] = red[0][g * 2 + mt][v][lane] + red[1][g * 2 + mt][v][lane]
             + red[2][g * 2 + mt][v][lane] + red[3][g * 2 + mt][v][lane];
      const float i_t = hsig(s[0] + biI);
      const float f_t = hsig(s[1] + biF);
      const float g_t = tanhf(s[2] + biC);
      const float o_t = hsig(s[3] + biO);
      const size_t cidx = (size_t)b * H_ + h;
      const float cn = f_t * cbuf[cidx] + i_t * g_t;
      cbuf[cidx] = cn;
      const size_t xidx = ((size_t)b * T_ + t) * H_ + h;
      const float hv = o_t * cn + x[xidx];
      out[xidx]  = hv;
      hcur[cidx] = (__bf16)hv;
    }
  }
}

// ---------------- K2: attention scores h @ mem^T ----------------
// grid: 32 blocks x 1 wave; each block = one 16-wide N tile, both M tiles.
__global__ __launch_bounds__(32) void k2_scores(
    const __bf16* __restrict__ hcur,   // [B,H] bf16
    const __bf16* __restrict__ membf,  // [N,H] bf16 (== B^T for this GEMM)
    float* __restrict__ scores) {      // [B,N] fp32
  const int nt = blockIdx.x;
  const int lane = threadIdx.x;
  v8f acc0 = {}, acc1 = {};
  for (int ks = 0; ks < 32; ++ks) {
    const int kk = ks * 32;
    v16bf a0 = load_a16(hcur, H_, 0,  kk, lane);
    v16bf a1 = load_a16(hcur, H_, 16, kk, lane);
    v16bf bb = load_b16(membf, H_, nt * 16, kk, lane);
    acc0 = wmma_bf16(a0, bb, acc0);
    acc1 = wmma_bf16(a1, bb, acc1);
  }
  const int half = lane >> 4, cc = lane & 15;
  const int n = nt * 16 + cc;
  #pragma unroll
  for (int v = 0; v < 8; ++v) {
    scores[(size_t)(v + 8 * half) * N_ + n]      = acc0[v];
    scores[(size_t)(16 + v + 8 * half) * N_ + n] = acc1[v];
  }
}

// ---------------- K3: softmax + r = attn @ mem ----------------
// grid: 16 blocks x 8 waves = 128 waves = 128 output tiles (2 x 64).
__global__ __launch_bounds__(256) void k3_attn(
    const float*  __restrict__ scores, // [B,N] fp32
    const __bf16* __restrict__ memT,   // [H,N] bf16 (== B^T for this GEMM)
    __bf16* __restrict__ rbf) {        // [B,H] bf16
  __shared__ __bf16 ebf[32][512];      // unnormalized exp(scores - max)
  __shared__ float  pred[32][8];
  __shared__ float  rowsum[32];
  const int tid = threadIdx.x;
  const int row = tid >> 3, seg = tid & 7;   // 8 threads per batch row

  float mx = -3.0e38f;
  for (int j = 0; j < 64; ++j)
    mx = fmaxf(mx, scores[(size_t)row * N_ + seg * 64 + j]);
  pred[row][seg] = mx;
  __syncthreads();
  if (seg == 0) {
    float m = pred[row][0];
    #pragma unroll
    for (int k = 1; k < 8; ++k) m = fmaxf(m, pred[row][k]);
    rowsum[row] = m;                   // temporarily holds row max
  }
  __syncthreads();
  const float rm = rowsum[row];
  float ssum = 0.0f;
  for (int j = 0; j < 64; ++j) {
    const float e = __expf(scores[(size_t)row * N_ + seg * 64 + j] - rm);
    ebf[row][seg * 64 + j] = (__bf16)e;
    ssum += e;
  }
  __syncthreads();                     // everyone has read rm
  pred[row][seg] = ssum;
  __syncthreads();
  if (seg == 0) {
    float s = 0.0f;
    #pragma unroll
    for (int k = 0; k < 8; ++k) s += pred[row][k];
    rowsum[row] = s;
  }
  __syncthreads();

  // GEMM: A (bf16, from LDS) [32,512] x memT^T -> r [32,1024]
  const int w = tid >> 5, lane = tid & 31;
  const int tile = blockIdx.x * 8 + w;       // 0..127
  const int mt = tile >> 6, ht = tile & 63;
  const __bf16* ap = &ebf[0][0];
  v8f acc = {};
  for (int ks = 0; ks < 16; ++ks) {
    const int kk = ks * 32;
    v16bf a  = load_a16(ap, 512, mt * 16, kk, lane);   // ds_load_b128 path
    v16bf bb = load_b16(memT, N_, ht * 16, kk, lane);
    acc = wmma_bf16(a, bb, acc);
  }
  const int half = lane >> 4, cc = lane & 15;
  const int h = ht * 16 + cc;
  #pragma unroll
  for (int v = 0; v < 8; ++v) {
    const int b = mt * 16 + v + 8 * half;
    rbf[(size_t)b * H_ + h] = (__bf16)(acc[v] / rowsum[b]);
  }
}

// ---------------- host launch ----------------
extern "C" void kernel_launch(void* const* d_in, const int* in_sizes, int n_in,
                              void* d_out, int out_size, void* d_ws, size_t ws_size,
                              hipStream_t stream) {
  (void)in_sizes; (void)n_in; (void)out_size; (void)ws_size;
  const float* x   = (const float*)d_in[0];
  const float* h0  = (const float*)d_in[1];
  const float* r0  = (const float*)d_in[2];
  const float* c0  = (const float*)d_in[3];
  const float* mem = (const float*)d_in[4];
  const float* WI  = (const float*)d_in[5];
  const float* bI  = (const float*)d_in[6];
  const float* WF  = (const float*)d_in[7];
  const float* bF  = (const float*)d_in[8];
  const float* WC  = (const float*)d_in[9];
  const float* bC  = (const float*)d_in[10];
  const float* WO  = (const float*)d_in[11];
  const float* bO  = (const float*)d_in[12];
  float* out = (float*)d_out;

  // workspace carve (~61.2 MB)
  char* p = (char*)d_ws;
  auto carve = [&](size_t bytes) -> char* {
    char* r = p; p += (bytes + 255) & ~(size_t)255; return r;
  };
  __bf16* wt     = (__bf16*)carve((size_t)4 * H_ * K3H_ * 2);
  __bf16* membf  = (__bf16*)carve((size_t)N_ * H_ * 2);
  __bf16* memT   = (__bf16*)carve((size_t)H_ * N_ * 2);
  __bf16* xbf    = (__bf16*)carve((size_t)B_ * T_ * H_ * 2);
  __bf16* hA     = (__bf16*)carve((size_t)B_ * H_ * 2);
  __bf16* hB     = (__bf16*)carve((size_t)B_ * H_ * 2);
  __bf16* rbf    = (__bf16*)carve((size_t)B_ * H_ * 2);
  float*  cbuf   = (float*) carve((size_t)B_ * H_ * 4);
  float*  scores = (float*) carve((size_t)B_ * N_ * 4);

  // prologue: bf16 conversion / weight transposes (L2-resident thereafter)
  const dim3 tb(32, 8);
  transpose_cvt_k<<<dim3(H_ / 32, K3H_ / 32), tb, 0, stream>>>(WI, wt + 0ull * H_ * K3H_, K3H_, H_);
  transpose_cvt_k<<<dim3(H_ / 32, K3H_ / 32), tb, 0, stream>>>(WF, wt + 1ull * H_ * K3H_, K3H_, H_);
  transpose_cvt_k<<<dim3(H_ / 32, K3H_ / 32), tb, 0, stream>>>(WC, wt + 2ull * H_ * K3H_, K3H_, H_);
  transpose_cvt_k<<<dim3(H_ / 32, K3H_ / 32), tb, 0, stream>>>(WO, wt + 3ull * H_ * K3H_, K3H_, H_);
  transpose_cvt_k<<<dim3(H_ / 32, N_ / 32), tb, 0, stream>>>(mem, memT, N_, H_);
  {
    int n = N_ * H_;
    cvt_bf16_k<<<(n + 255) / 256, 256, 0, stream>>>(mem, membf, n);
  }
  {
    int n = B_ * T_ * H_;
    cvt_bf16_k<<<(n + 255) / 256, 256, 0, stream>>>(x, xbf, n);
  }
  {
    int n = B_ * H_;
    cvt_bf16_k<<<(n + 255) / 256, 256, 0, stream>>>(h0, hA, n);
    cvt_bf16_k<<<(n + 255) / 256, 256, 0, stream>>>(r0, rbf, n);
    copy_f32_k<<<(n + 255) / 256, 256, 0, stream>>>(c0, cbuf, n);
  }

  // sequential recurrence: 3 graph-captured launches per timestep
  for (int t = 0; t < T_; ++t) {
    const __bf16* hprev = (t & 1) ? hB : hA;
    __bf16*       hcur  = (t & 1) ? hA : hB;
    k1_gates<<<64, 128, 0, stream>>>(wt, hprev, rbf, xbf, x,
                                     bI, bF, bC, bO, cbuf, out, hcur, t);
    k2_scores<<<32, 32, 0, stream>>>(hcur, membf, scores);
    k3_attn<<<16, 256, 0, stream>>>(scores, memT, rbf);
  }
}
